// HybridQCNN_78288663871857
// MI455X (gfx1250) — matrix operand order repaired and checked by reading, not compiled, not run on hardware
//
#include <hip/hip_runtime.h>
#include <hip/hip_bf16.h>
#include <cstddef>

// ---------------------------------------------------------------------------
// Types / feature detection
// ---------------------------------------------------------------------------
typedef _Float16 half_t;
typedef __attribute__((ext_vector_type(16))) _Float16 v16h;
typedef __attribute__((ext_vector_type(8)))  float    v8f;
typedef __attribute__((ext_vector_type(4)))  int      v4i_t;

union U32H2 { unsigned u; half_t h[2]; };

#define AS1 __attribute__((address_space(1)))
#define AS3 __attribute__((address_space(3)))

#if defined(__has_builtin)
#if __has_builtin(__builtin_amdgcn_global_load_async_to_lds_b128)
#define HAVE_ASYNC_LDS 1
#endif
#if __has_builtin(__builtin_amdgcn_s_wait_asynccnt)
#define ASYNC_WAIT0() __builtin_amdgcn_s_wait_asynccnt(0)
#endif
#endif
#if defined(HAVE_ASYNC_LDS) && !defined(ASYNC_WAIT0)
#define ASYNC_WAIT0() asm volatile("s_wait_asynccnt 0x0" ::: "memory")
#endif

__device__ __forceinline__ float gelu_exact(float x) {
    return 0.5f * x * (1.0f + erff(x * 0.70710678118654752f));
}

// ---------------------------------------------------------------------------
// Kernel 1: first conv 3x3 stride-2, 3->32, pad 1, 224 -> 112.
// x: [B][3][224][224] fp32, W: [32][3][3][3] fp32, out: [32][B][112][112] f16
// ---------------------------------------------------------------------------
__global__ __launch_bounds__(256)
void conv0_kernel(const float* __restrict__ x, const float* __restrict__ W,
                  half_t* __restrict__ out, int Bn) {
    size_t idx = (size_t)blockIdx.x * blockDim.x + threadIdx.x;
    size_t total = (size_t)32 * Bn * 112 * 112;
    if (idx >= total) return;
    int ow = idx % 112; size_t t = idx / 112;
    int oh = t % 112;   t /= 112;
    int b  = t % Bn;
    int co = t / Bn;
    float acc = 0.f;
    for (int ci = 0; ci < 3; ++ci) {
        for (int kh = 0; kh < 3; ++kh) {
            int ih = 2 * oh - 1 + kh;
            if (ih < 0 || ih >= 224) continue;
            for (int kw = 0; kw < 3; ++kw) {
                int iw = 2 * ow - 1 + kw;
                if (iw < 0 || iw >= 224) continue;
                float xv = x[(((size_t)b * 3 + ci) * 224 + ih) * 224 + iw];
                float wv = W[((co * 3 + ci) * 3 + kh) * 3 + kw];
                acc += xv * wv;
            }
        }
    }
    out[idx] = (half_t)acc;
}

// ---------------------------------------------------------------------------
// Kernel 2: WMMA GEMM for pointwise (1x1) convolutions.
//   C[M][N] (f16) = A[M][K] (fp32 weights, cvt f16) * B[K][N] (f16 activations)
// 256 threads = 8 waves; block tile 64(M) x 64(N), K step 32.
// Wave w: rows [ (w>>1)*16 .. +16 ), cols [ (w&1)*32 .. +32 ) -> two 16x16 accums.
//
// Shape guarantees (MobileNetV2 on this harness): K % 8 == 0, N % 16 == 0,
// so staging uses clamped aligned 128-bit accesses + zero masks, branch-free.
//
// gfx1250 ASYNCcnt path: the B tile (pure copy, no conversion) is staged with
// GLOBAL_LOAD_ASYNC_TO_LDS_B128 into natural [k][n] layout; fragments are
// gathered with strided DS reads + edge masks. The A tile stays on the VALU
// path (needs fp32->f16 conversion).
// ---------------------------------------------------------------------------
#define GEMM_BM 64
#define GEMM_BN 64
#define GEMM_BK 32
#define LDS_PAD 8

__global__ __launch_bounds__(256)
void gemm_wmma_kernel(const float* __restrict__ A, const half_t* __restrict__ Bm,
                      half_t* __restrict__ C, int M, long long N, int K) {
    __shared__ half_t As[GEMM_BM][GEMM_BK + LDS_PAD];   // [m][k]
#ifdef HAVE_ASYNC_LDS
    __shared__ half_t Braw[GEMM_BK][GEMM_BN];           // [k][n] natural layout
#else
    __shared__ half_t Bs[GEMM_BN][GEMM_BK + LDS_PAD];   // [n][k] transposed stage
#endif

    const long long n0 = (long long)blockIdx.x * GEMM_BN;
    const int       m0 = blockIdx.y * GEMM_BM;
    const int tid  = threadIdx.x;
    const int lane = tid & 31;
    const int wave = tid >> 5;
    const int wm = (wave >> 1) * 16;   // 0,16,32,48
    const int wn = (wave & 1) * 32;    // 0,32

    v8f acc0 = {};
    v8f acc1 = {};

    // Per-thread staging coordinates (fixed across K loop)
    const int aM0 = (tid + 0)   >> 3;        // A chunk 0 row   (0..31)
    const int aK0 = ((tid + 0)  & 7) * 4;    // A chunk 0 k     (0..28)
    const int aM1 = (tid + 256) >> 3;        // A chunk 1 row   (32..63)
    const int bK  = tid >> 3;                // B row k (0..31)
    const int bN  = (tid & 7) * 8;           // B col chunk (0..56)

    for (int k0 = 0; k0 < K; k0 += GEMM_BK) {
#ifdef HAVE_ASYNC_LDS
        // ---- issue async B-tile copy first: global [k][n] -> LDS [k][n] ----
        {
            long long gk = k0 + bK;
            long long gn = n0 + bN;
            long long rk = (gk < K) ? gk : (K - 1);
            long long rn = (gn < N) ? gn : (N - 8);    // N % 16 == 0 -> aligned
            __builtin_amdgcn_global_load_async_to_lds_b128(
                (AS1 v4i_t*)(Bm + (size_t)rk * N + rn),
                (AS3 v4i_t*)&Braw[bK][bN],
                0, 0);
        }
#endif
        // ---- stage A tile: two aligned float4 chunks per thread, clamped ----
        {
            #pragma unroll
            for (int i = 0; i < 2; ++i) {
                int m  = (i == 0) ? aM0 : aM1;
                int kc = aK0;
                int gm = m0 + m;
                int gk = k0 + kc;
                int rm = (gm < M) ? gm : (M - 1);
                int rk = (gk < K) ? gk : (K - 4);      // K % 8 == 0 -> aligned
                float msk = ((gm < M) && (gk < K)) ? 1.f : 0.f;
                const float4 v = *reinterpret_cast<const float4*>(A + (size_t)rm * K + rk);
                As[m][kc + 0] = (half_t)(v.x * msk);
                As[m][kc + 1] = (half_t)(v.y * msk);
                As[m][kc + 2] = (half_t)(v.z * msk);
                As[m][kc + 3] = (half_t)(v.w * msk);
            }
        }
#ifndef HAVE_ASYNC_LDS
        // ---- stage B tile: one aligned uint4 (8 f16) per thread, clamped,
        //      scatter-transposed into Bs[n][k] ----
        {
            long long gk = k0 + bK;
            long long gn = n0 + bN;
            long long rk = (gk < K) ? gk : (K - 1);
            long long rn = (gn < N) ? gn : (N - 8);    // N % 16 == 0 -> aligned
            bool valid = (gk < K) && (gn < N);
            const uint4 raw = *reinterpret_cast<const uint4*>(Bm + (size_t)rk * N + rn);
            unsigned uu[4] = { raw.x, raw.y, raw.z, raw.w };
            #pragma unroll
            for (int j = 0; j < 4; ++j) {
                U32H2 p; p.u = uu[j];
                Bs[bN + 2 * j + 0][bK] = valid ? p.h[0] : (half_t)0.f;
                Bs[bN + 2 * j + 1][bK] = valid ? p.h[1] : (half_t)0.f;
            }
        }
#endif
        // CDNA5 speculative prefetch of next K tile (global_prefetch_b8)
        if (k0 + GEMM_BK < K) {
            __builtin_prefetch(&A[(size_t)(m0 + (tid >> 2)) * K + (k0 + GEMM_BK)], 0, 1);
            __builtin_prefetch(&Bm[(size_t)(k0 + GEMM_BK + (tid >> 6)) * N + n0], 0, 1);
        }
#ifdef HAVE_ASYNC_LDS
        ASYNC_WAIT0();          // this wave's async deposits have landed
#endif
        __syncthreads();        // all waves' deposits visible

        // ---- A fragment: lane<16 -> K {0..15}; lane>=16 -> K {16..31}
        v16h af;
        {
            int row = wm + (lane & 15);
            int kh  = (lane < 16) ? 0 : 16;
            #pragma unroll
            for (int j = 0; j < 8; ++j) {
                int kk = kh + 2 * j;
                af[2 * j]     = As[row][kk];
                af[2 * j + 1] = As[row][kk + 1];
            }
        }
        // ---- B fragments + WMMA (two 16-col halves) ----
        {
            int kb = (lane < 16) ? 0 : 16;
            v16h bf0, bf1;
            int col0 = wn + (lane & 15);
            int col1 = wn + 16 + (lane & 15);
#ifdef HAVE_ASYNC_LDS
            bool ok0 = (n0 + col0) < N;
            bool ok1 = (n0 + col1) < N;
            #pragma unroll
            for (int j = 0; j < 16; ++j) {
                int kk = kb + j;
                bool krow = (k0 + kk) < K;
                half_t va = Braw[kk][col0];
                half_t vb = Braw[kk][col1];
                bf0[j] = (ok0 && krow) ? va : (half_t)0.f;
                bf1[j] = (ok1 && krow) ? vb : (half_t)0.f;
            }
#else
            #pragma unroll
            for (int j = 0; j < 8; ++j) {
                bf0[2 * j]     = Bs[col0][kb + 2 * j];
                bf0[2 * j + 1] = Bs[col0][kb + 2 * j + 1];
                bf1[2 * j]     = Bs[col1][kb + 2 * j];
                bf1[2 * j + 1] = Bs[col1][kb + 2 * j + 1];
            }
#endif
            acc0 = __builtin_amdgcn_wmma_f32_16x16x32_f16(false, af, false, bf0,
                                                          (short)0, acc0, false, false);
            acc1 = __builtin_amdgcn_wmma_f32_16x16x32_f16(false, af, false, bf1,
                                                          (short)0, acc1, false, false);
        }
        __syncthreads();
    }

    // ---- store D (f32 C/D layout: lane%16 = N col; vgpr r = M row r (+8 hi half))
    const int ncol = lane & 15;
    const int mrow = (lane < 16) ? 0 : 8;
    #pragma unroll
    for (int r = 0; r < 8; ++r) {
        int m = m0 + wm + mrow + r;
        long long na = n0 + wn + ncol;
        if (m < M && na < N) C[(size_t)m * N + na] = (half_t)acc0[r];
        long long nb = n0 + wn + 16 + ncol;
        if (m < M && nb < N) C[(size_t)m * N + nb] = (half_t)acc1[r];
    }
}

// ---------------------------------------------------------------------------
// Kernel 3: depthwise 3x3 conv, pad 1, stride s.
// X: [C][B][H][W] f16, W: [C][1][3][3] fp32, Y: [C][B][Ho][Wo] f16
// ---------------------------------------------------------------------------
__global__ __launch_bounds__(256)
void dwconv_kernel(const half_t* __restrict__ X, const float* __restrict__ W,
                   half_t* __restrict__ Y, int C, int Bn, int Hin, int Win,
                   int Ho, int Wo, int stride) {
    size_t idx = (size_t)blockIdx.x * blockDim.x + threadIdx.x;
    size_t total = (size_t)C * Bn * Ho * Wo;
    if (idx >= total) return;
    int wo = idx % Wo; size_t t = idx / Wo;
    int ho = t % Ho;   t /= Ho;
    int b  = t % Bn;
    int c  = t / Bn;
    const float* wk = W + (size_t)c * 9;
    float acc = 0.f;
    #pragma unroll
    for (int kh = 0; kh < 3; ++kh) {
        int ih = ho * stride - 1 + kh;
        if (ih < 0 || ih >= Hin) continue;
        #pragma unroll
        for (int kw = 0; kw < 3; ++kw) {
            int iw = wo * stride - 1 + kw;
            if (iw < 0 || iw >= Win) continue;
            acc += (float)X[(((size_t)c * Bn + b) * Hin + ih) * Win + iw] * wk[kh * 3 + kw];
        }
    }
    Y[idx] = (half_t)acc;
}

// ---------------------------------------------------------------------------
// Kernel 4: BatchNorm statistics (training mode): per-channel mean & rstd.
// X: [C][N] f16, stats: [C][2] fp32 (mean, rsqrt(var+eps)). One block / channel.
// ---------------------------------------------------------------------------
__global__ __launch_bounds__(256)
void bn_stats_kernel(const half_t* __restrict__ X, float* __restrict__ stats,
                     long long N) {
    const int c = blockIdx.x;
    const half_t* row = X + (size_t)c * N;
    float s = 0.f, ss = 0.f;
    for (long long i = threadIdx.x; i < N; i += blockDim.x) {
        float v = (float)row[i];
        s += v; ss += v * v;
    }
    __shared__ float r1[256], r2[256];
    int tid = threadIdx.x;
    r1[tid] = s; r2[tid] = ss;
    __syncthreads();
    for (int off = 128; off > 0; off >>= 1) {
        if (tid < off) { r1[tid] += r1[tid + off]; r2[tid] += r2[tid + off]; }
        __syncthreads();
    }
    if (tid == 0) {
        float m = r1[0] / (float)N;
        float var = r2[0] / (float)N - m * m;
        stats[2 * c]     = m;
        stats[2 * c + 1] = rsqrtf(var + 1e-5f);
    }
}

// ---------------------------------------------------------------------------
// Kernel 5: BN apply + optional ReLU6: y = clip((x-m)*rs*g + b, 0, 6)
// grid = (ceil(N/256), C)
// ---------------------------------------------------------------------------
__global__ __launch_bounds__(256)
void bn_apply_kernel(const half_t* __restrict__ X, half_t* __restrict__ Y,
                     const float* __restrict__ stats, const float* __restrict__ g,
                     const float* __restrict__ be, long long N, int relu6) {
    int c = blockIdx.y;
    long long i = (long long)blockIdx.x * blockDim.x + threadIdx.x;
    if (i >= N) return;
    float m  = stats[2 * c];
    float rs = stats[2 * c + 1];
    float v = ((float)X[(size_t)c * N + i] - m) * rs * g[c] + be[c];
    if (relu6) v = fminf(fmaxf(v, 0.f), 6.f);
    Y[(size_t)c * N + i] = (half_t)v;
}

// ---------------------------------------------------------------------------
// Kernel 6: residual add (in place): y += x, f16 with f32 math.
// ---------------------------------------------------------------------------
__global__ __launch_bounds__(256)
void add_kernel(half_t* __restrict__ y, const half_t* __restrict__ x, size_t n) {
    size_t i = (size_t)blockIdx.x * blockDim.x + threadIdx.x;
    if (i >= n) return;
    y[i] = (half_t)((float)y[i] + (float)x[i]);
}

// ---------------------------------------------------------------------------
// Kernel 7: global average pool over 7x7: [1280][B][49] f16 -> feat [B][1280] f32
// ---------------------------------------------------------------------------
__global__ __launch_bounds__(256)
void avgpool_kernel(const half_t* __restrict__ X, float* __restrict__ feat,
                    int Bn, int C, int sp) {
    int idx = blockIdx.x * blockDim.x + threadIdx.x;
    if (idx >= Bn * C) return;
    int b = idx / C;
    int c = idx % C;
    const half_t* p = X + ((size_t)c * Bn + b) * sp;
    float s = 0.f;
    for (int i = 0; i < sp; ++i) s += (float)p[i];
    feat[(size_t)b * C + c] = s / (float)sp;
}

// ---------------------------------------------------------------------------
// 4-qubit statevector simulator (device, single lane). Wire 0 = MSB of index.
// ---------------------------------------------------------------------------
struct cpx { float re, im; };
__device__ __forceinline__ cpx cmul(cpx a, cpx b) {
    cpx r; r.re = a.re * b.re - a.im * b.im; r.im = a.re * b.im + a.im * b.re; return r;
}
__device__ __forceinline__ cpx cadd(cpx a, cpx b) { cpx r; r.re = a.re + b.re; r.im = a.im + b.im; return r; }

__device__ void q_apply1(cpx* st, const cpx* U, int w) {
    int step = 1 << (3 - w);
    for (int i = 0; i < 16; ++i) {
        if (i & step) continue;
        cpx s0 = st[i], s1 = st[i | step];
        st[i]        = cadd(cmul(U[0], s0), cmul(U[1], s1));
        st[i | step] = cadd(cmul(U[2], s0), cmul(U[3], s1));
    }
}
__device__ void q_apply2(cpx* st, const cpx* U, int w0, int w1) {
    int m0 = 1 << (3 - w0), m1 = 1 << (3 - w1);
    for (int i = 0; i < 16; ++i) {
        if ((i & m0) || (i & m1)) continue;
        int id[4] = { i, i | m1, i | m0, i | m0 | m1 };   // vec index = 2*bit(w0)+bit(w1)
        cpx v[4];
        for (int a = 0; a < 4; ++a) v[a] = st[id[a]];
        for (int r = 0; r < 4; ++r) {
            cpx acc; acc.re = 0.f; acc.im = 0.f;
            for (int k = 0; k < 4; ++k) acc = cadd(acc, cmul(U[r * 4 + k], v[k]));
            st[id[r]] = acc;
        }
    }
}
__device__ void mk_u3(cpx* U, float t, float p, float l) {
    float ct = cosf(0.5f * t), sn = sinf(0.5f * t);
    U[0].re = ct;                U[0].im = 0.f;
    U[1].re = -cosf(l) * sn;     U[1].im = -sinf(l) * sn;
    U[2].re =  cosf(p) * sn;     U[2].im =  sinf(p) * sn;
    U[3].re =  cosf(p + l) * ct; U[3].im =  sinf(p + l) * ct;
}
__device__ void mk_ident4(cpx* U) {
    for (int i = 0; i < 16; ++i) { U[i].re = 0.f; U[i].im = 0.f; }
    U[0].re = 1.f; U[5].re = 1.f; U[10].re = 1.f; U[15].re = 1.f;
}
__device__ void mk_cnot(cpx* U) {   // control = first wire
    mk_ident4(U);
    U[10].re = 0.f; U[11].re = 1.f; U[14].re = 1.f; U[15].re = 0.f;
}
__device__ void mk_crz(cpx* U, float phi) {
    mk_ident4(U);
    float c = cosf(0.5f * phi), s = sinf(0.5f * phi);
    U[10].re = c; U[10].im = -s;     // e^{-i phi/2}
    U[15].re = c; U[15].im =  s;     // conj
}
__device__ void mk_crx(cpx* U, float t) {
    mk_ident4(U);
    float c = cosf(0.5f * t), s = sinf(0.5f * t);
    U[10].re = c;   U[10].im = 0.f;
    U[11].re = 0.f; U[11].im = -s;
    U[14].re = 0.f; U[14].im = -s;
    U[15].re = c;   U[15].im = 0.f;
}
__device__ void q_conv(cpx* st, const float* w, int w0, int w1) {
    cpx U[16], u[4];
    mk_u3(u, w[0], w[1], w[2]);   q_apply1(st, u, w0);
    mk_u3(u, w[3], w[4], w[5]);   q_apply1(st, u, w1);
    mk_cnot(U);                   q_apply2(st, U, w0, w1);
    mk_u3(u, w[6], w[7], w[8]);   q_apply1(st, u, w0);
    mk_u3(u, w[9], w[10], w[11]); q_apply1(st, u, w1);
    q_apply2(st, U, w1, w0);
    mk_u3(u, w[12], w[13], w[14]); q_apply1(st, u, w0);
}
__device__ void q_pool(cpx* st, const float* w, int w0, int w1) {
    cpx U[16], X[4];
    X[0].re = 0.f; X[0].im = 0.f; X[1].re = 1.f; X[1].im = 0.f;
    X[2].re = 1.f; X[2].im = 0.f; X[3].re = 0.f; X[3].im = 0.f;
    mk_crz(U, w[0]); q_apply2(st, U, w0, w1);
    q_apply1(st, X, w0);
    mk_crx(U, w[1]); q_apply2(st, U, w0, w1);
    q_apply1(st, X, w0);
}
__device__ float qcnn_sim(const float* red, const float* cw, const float* pw) {
    float cq[4], sq[4];
    for (int q = 0; q < 4; ++q) {
        float a = red[q] * 3.14159265358979323846f * 0.5f;  // angle = red*pi, use angle/2
        cq[q] = cosf(a); sq[q] = sinf(a);
    }
    cpx st[16];
    for (int i = 0; i < 16; ++i) {
        float amp = (((i >> 3) & 1) ? sq[0] : cq[0]) * (((i >> 2) & 1) ? sq[1] : cq[1]) *
                    (((i >> 1) & 1) ? sq[2] : cq[2]) * ((i & 1) ? sq[3] : cq[3]);
        st[i].re = amp; st[i].im = 0.f;
    }
    int active[4] = {0, 1, 2, 3};
    int na = 4;
    for (int layer = 0; layer < 2; ++layer) {
        for (int i = 0; i < na; ++i)
            q_conv(st, cw + (layer * 4 + i) * 15, active[i], active[(i + 1) % na]);
        int nn = 0;
        for (int j = 0; j + 1 < na; j += 2) {
            q_pool(st, pw + (layer * 2 + j / 2) * 2, active[j], active[j + 1]);
            active[nn++] = active[j + 1];
        }
        na = nn;
    }
    int bit = 3 - active[0];
    float e = 0.f;
    for (int i = 0; i < 16; ++i) {
        float p = st[i].re * st[i].re + st[i].im * st[i].im;
        e += ((i >> bit) & 1) ? -p : p;
    }
    return e;
}

// ---------------------------------------------------------------------------
// Kernel 8: fused heads. One workgroup per batch element.
// feat[B][1280] -> reduced[4] -> {qcnn, bypass} -> classifier -> out[B][5]
// ---------------------------------------------------------------------------
__global__ __launch_bounds__(256)
void head_kernel(const float* __restrict__ feat,
                 const float* __restrict__ rW1, const float* __restrict__ rb1,
                 const float* __restrict__ rW2, const float* __restrict__ rb2,
                 const float* __restrict__ cw,  const float* __restrict__ pw,
                 const float* __restrict__ bW,  const float* __restrict__ bb,
                 const float* __restrict__ cW1, const float* __restrict__ cb1,
                 const float* __restrict__ cW2, const float* __restrict__ cb2,
                 float* __restrict__ out) {
    const int b = blockIdx.x, tid = threadIdx.x;
    __shared__ float fs[1280];
    __shared__ float h1[128];
    __shared__ float red[4];
    __shared__ float byp[32];
    __shared__ float hh[64];
    __shared__ float qout_s;

    for (int i = tid; i < 1280; i += 256) fs[i] = feat[(size_t)b * 1280 + i];
    __syncthreads();

    if (tid < 128) {
        float a = rb1[tid];
        for (int k = 0; k < 1280; ++k) a += fs[k] * rW1[(size_t)k * 128 + tid];
        h1[tid] = gelu_exact(a);
    }
    __syncthreads();

    if (tid < 4) {
        float a = rb2[tid];
        for (int j = 0; j < 128; ++j) a += h1[j] * rW2[j * 4 + tid];
        red[tid] = tanhf(a);
    }
    __syncthreads();

    if (tid < 32) {
        float a = bb[tid];
        for (int q = 0; q < 4; ++q) a += red[q] * bW[q * 32 + tid];
        byp[tid] = gelu_exact(a);
    }
    if (tid == 32) qout_s = qcnn_sim(red, cw, pw);
    __syncthreads();

    if (tid < 64) {
        float a = cb1[tid] + qout_s * cW1[tid];           // combined[0] = q_out
        for (int i = 0; i < 32; ++i) a += byp[i] * cW1[(i + 1) * 64 + tid];
        hh[tid] = gelu_exact(a);
    }
    __syncthreads();

    if (tid < 5) {
        float a = cb2[tid];
        for (int j = 0; j < 64; ++j) a += hh[j] * cW2[j * 5 + tid];
        out[(size_t)b * 5 + tid] = a;
    }
}

// ---------------------------------------------------------------------------
// Host orchestration
// ---------------------------------------------------------------------------
extern "C" void kernel_launch(void* const* d_in, const int* in_sizes, int n_in,
                              void* d_out, int out_size, void* d_ws, size_t ws_size,
                              hipStream_t stream) {
    (void)in_sizes; (void)n_in; (void)out_size; (void)ws_size;
    const int Bn = 32;
    const float* x = (const float*)d_in[0];

    // ---- carve workspace ----
    char* base = (char*)d_ws;
    size_t off = 0;
    auto carve = [&](size_t bytes) -> void* {
        void* p = base + off;
        off = (off + bytes + 255) & ~(size_t)255;
        return p;
    };
    const size_t MAXE = (size_t)96 * Bn * 112 * 112;    // largest activation tensor
    half_t* convbuf = (half_t*)carve(MAXE * sizeof(half_t));
    half_t* act[4];
    for (int i = 0; i < 4; ++i) act[i] = (half_t*)carve(MAXE * sizeof(half_t));
    float* stats = (float*)carve((size_t)2 * 1280 * sizeof(float));
    float* feat  = (float*)carve((size_t)Bn * 1280 * sizeof(float));

    int pi = 1;   // d_in index cursor into backbone_params (W, gamma, beta per conv)
    auto conv_params = [&](const float** W, const float** g, const float** b) {
        *W = (const float*)d_in[pi];
        *g = (const float*)d_in[pi + 1];
        *b = (const float*)d_in[pi + 2];
        pi += 3;
    };
    auto bn = [&](const half_t* pre, half_t* post, int C, long long N,
                  const float* g, const float* be, bool relu) {
        bn_stats_kernel<<<C, 256, 0, stream>>>(pre, stats, N);
        dim3 grid((unsigned)((N + 255) / 256), (unsigned)C);
        bn_apply_kernel<<<grid, 256, 0, stream>>>(pre, post, stats, g, be, N, relu ? 1 : 0);
    };
    auto pointwise = [&](const half_t* in, half_t* outAct, int ic, int oc, int H, bool relu) {
        const float *W, *g, *be; conv_params(&W, &g, &be);
        long long N = (long long)Bn * H * H;
        dim3 grid((unsigned)((N + GEMM_BN - 1) / GEMM_BN),
                  (unsigned)((oc + GEMM_BM - 1) / GEMM_BM));
        gemm_wmma_kernel<<<grid, 256, 0, stream>>>(W, in, convbuf, oc, N, ic);
        bn(convbuf, outAct, oc, N, g, be, relu);
    };
    auto depthwise = [&](const half_t* in, half_t* outAct, int C, int H, int stride, int* HoOut) {
        const float *W, *g, *be; conv_params(&W, &g, &be);
        int Ho = (H + 2 - 3) / stride + 1;
        size_t total = (size_t)C * Bn * Ho * Ho;
        dwconv_kernel<<<(unsigned)((total + 255) / 256), 256, 0, stream>>>(
            in, W, convbuf, C, Bn, H, H, Ho, Ho, stride);
        bn(convbuf, outAct, C, (long long)Bn * Ho * Ho, g, be, true);
        *HoOut = Ho;
    };

    // ---- conv0: 3->32, s2, 224->112 ----
    {
        const float *W, *g, *be; conv_params(&W, &g, &be);
        size_t total = (size_t)32 * Bn * 112 * 112;
        conv0_kernel<<<(unsigned)((total + 255) / 256), 256, 0, stream>>>(x, W, convbuf, Bn);
        bn(convbuf, act[0], 32, (long long)Bn * 112 * 112, g, be, true);
    }

    static const int cfg[7][4] = {{1,16,1,1},{6,24,2,2},{6,32,3,2},{6,64,4,2},
                                  {6,96,3,1},{6,160,3,2},{6,320,1,1}};
    int cur = 0, H = 112, in_c = 32;
    for (int ci = 0; ci < 7; ++ci) {
        int t = cfg[ci][0], c = cfg[ci][1], n = cfg[ci][2], s = cfg[ci][3];
        for (int k = 0; k < n; ++k) {
            int stride = (k == 0) ? s : 1;
            int hid = in_c * t;
            int inSlot = cur;
            const half_t* hin = act[inSlot];
            int slot = inSlot;
            if (t != 1) {
                slot = (slot + 1) & 3;
                pointwise(hin, act[slot], in_c, hid, H, true);   // expand 1x1 + ReLU6
                hin = act[slot];
            }
            int Ho;
            int slot2 = (slot + 1) & 3;
            depthwise(hin, act[slot2], hid, H, stride, &Ho);     // dw 3x3 + ReLU6
            int slot3 = (slot2 + 1) & 3;
            pointwise(act[slot2], act[slot3], hid, c, Ho, false); // project 1x1 (no act)
            if (stride == 1 && in_c == c) {
                size_t total = (size_t)c * Bn * Ho * Ho;
                add_kernel<<<(unsigned)((total + 255) / 256), 256, 0, stream>>>(
                    act[slot3], act[inSlot], total);
            }
            cur = slot3; H = Ho; in_c = c;
        }
    }

    // ---- final 1x1: in_c -> 1280 + ReLU6 ----
    {
        int slot = (cur + 1) & 3;
        pointwise(act[cur], act[slot], in_c, 1280, H, true);
        cur = slot;
    }

    // ---- global average pool -> feat [B][1280] ----
    {
        int total = Bn * 1280;
        avgpool_kernel<<<(total + 255) / 256, 256, 0, stream>>>(act[cur], feat, Bn, 1280, H * H);
    }

    // ---- fused heads (+ 4-qubit sim) ----
    const float* rW1 = (const float*)d_in[pi + 0];
    const float* rb1 = (const float*)d_in[pi + 1];
    const float* rW2 = (const float*)d_in[pi + 2];
    const float* rb2 = (const float*)d_in[pi + 3];
    const float* cw  = (const float*)d_in[pi + 4];
    const float* pw  = (const float*)d_in[pi + 5];
    const float* bW  = (const float*)d_in[pi + 6];
    const float* bb  = (const float*)d_in[pi + 7];
    const float* cW1 = (const float*)d_in[pi + 8];
    const float* cb1 = (const float*)d_in[pi + 9];
    const float* cW2 = (const float*)d_in[pi + 10];
    const float* cb2 = (const float*)d_in[pi + 11];
    head_kernel<<<Bn, 256, 0, stream>>>(feat, rW1, rb1, rW2, rb2, cw, pw,
                                        bW, bb, cW1, cb1, cW2, cb2, (float*)d_out);
}
